// Head_3702261809270
// MI455X (gfx1250) — matrix operand order repaired and checked by reading, not compiled, not run on hardware
//
#include <hip/hip_runtime.h>

// ---------------------------------------------------------------------------
// Types
// ---------------------------------------------------------------------------
typedef __attribute__((ext_vector_type(16))) __bf16 v16bf;
typedef __attribute__((ext_vector_type(8)))  __bf16 bf16x8;
typedef __attribute__((ext_vector_type(4)))  __bf16 bf16x4;
typedef __attribute__((ext_vector_type(8)))  float  v8f;
typedef __attribute__((ext_vector_type(4)))  float  f32x4;
typedef __attribute__((ext_vector_type(4)))  unsigned int u32x4;
typedef __attribute__((ext_vector_type(8)))  int         i32x8;
typedef __attribute__((ext_vector_type(4)))  int         i32x4;

#define D_MODEL  2048
#define D_FF     8192
#define HEAD_DIM 64
#define N_HEADS  128
#define TOKENS   8192   // B*T = 4*2048

// ---------------------------------------------------------------------------
// WMMA helper:  D(16x16,f32) = A(16x32,bf16) * B(32x16,bf16) + C
// ---------------------------------------------------------------------------
__device__ __forceinline__ v8f wmma_bf16(v16bf a, v16bf b, v8f c) {
  return __builtin_amdgcn_wmma_f32_16x16x32_bf16(
      /*neg_a=*/false, a, /*neg_b=*/false, b,
      /*c_mod=*/(short)0, c, /*reuse_a=*/false, /*reuse_b=*/false);
}

// A fragment (16x32 bf16, row-major source [*, ld], tile at (row0, k0)).
// Layout: lanes 0-15 -> M=lane, K=0..15 contiguous; lanes 16-31 -> K=16..31.
// Two 16B contiguous chunks per lane.
__device__ __forceinline__ v16bf load_frag_a(const __bf16* base, int ld,
                                             int row0, int k0, int lane) {
  const int l  = lane & 15;
  const int hi = lane >> 4;
  const __bf16* p = base + (size_t)(row0 + l) * ld + k0 + hi * 16;
  v16bf f;
  *(bf16x8*)&f         = *(const bf16x8*)p;        // K = base .. base+7
  *(((bf16x8*)&f) + 1) = *(const bf16x8*)(p + 8);  // K = base+8 .. base+15
  return f;
}

// B fragment (32x16 bf16) built from a row-major [N, K] matrix Wrow, so that
// B[k][n] = Wrow[col0+n][k0+k]. Layout used here (contiguous k, matching load_frag_a): lanes 0-15 hold K=0..15 (16
// contiguous bf16), lanes 16-31 hold K=16..31. One 32B chunk per lane.
__device__ __forceinline__ v16bf load_frag_b(const __bf16* base, int ld,
                                             int col0, int k0, int lane) {
  const int n  = lane & 15;
  const int hi = lane >> 4;
  const __bf16* p = base + (size_t)(col0 + n) * ld + k0 + hi * 16;
  return *(const v16bf*)p;
}

// ---------------------------------------------------------------------------
// f32 -> bf16 conversion (vectorized, n divisible by 4)
// ---------------------------------------------------------------------------
__global__ __launch_bounds__(256) void cvt_f32_to_bf16(
    const float* __restrict__ src, __bf16* __restrict__ dst, int n4) {
  int i = blockIdx.x * blockDim.x + threadIdx.x;
  if (i < n4) {
    f32x4 v = ((const f32x4*)src)[i];
    bf16x4 o;
    o.x = (__bf16)v.x; o.y = (__bf16)v.y;
    o.z = (__bf16)v.z; o.w = (__bf16)v.w;
    ((bf16x4*)dst)[i] = o;
  }
}

// ---------------------------------------------------------------------------
// GEMM:  C[M,N] = A[M,K] * Wrow[N,K]^T   (bf16 inputs, f32 accum)
// 256 threads = 8 waves; block tile 256x128; wave tile 64x64 (4x4 WMMA tiles)
// => 8 fragment loads (4 A + 4 B) per 16 WMMAs in the K loop.
// ---------------------------------------------------------------------------
template <bool OUT_BF16>
__global__ __launch_bounds__(256) void gemm_xwT(
    const __bf16* __restrict__ A, const __bf16* __restrict__ Wrow,
    void* __restrict__ Cv, int M, int N, int K) {
  const int lane = threadIdx.x & 31;
  const int wid  = threadIdx.x >> 5;
  const int row0 = blockIdx.y * 256 + (wid >> 1) * 64;  // 4 m-tiles of 16
  const int col0 = blockIdx.x * 128 + (wid & 1) * 64;   // 4 n-tiles of 16

  v8f acc[4][4];
#pragma unroll
  for (int mi = 0; mi < 4; ++mi)
#pragma unroll
    for (int t = 0; t < 4; ++t) acc[mi][t] = (v8f){};

#pragma unroll 2
  for (int k0 = 0; k0 < K; k0 += 32) {
    // Prefetch the streaming operands ~8 K-steps ahead (global_prefetch_b8).
    if (k0 + 256 < K) {
      __builtin_prefetch(A    + (size_t)(row0 + (lane & 15)) * K + k0 + 256, 0, 0);
      __builtin_prefetch(Wrow + (size_t)(col0 + (lane & 15)) * K + k0 + 256, 0, 0);
    }
    v16bf a[4], b[4];
#pragma unroll
    for (int i = 0; i < 4; ++i)
      a[i] = load_frag_a(A, K, row0 + i * 16, k0, lane);
#pragma unroll
    for (int i = 0; i < 4; ++i)
      b[i] = load_frag_b(Wrow, K, col0 + i * 16, k0, lane);
#pragma unroll
    for (int mi = 0; mi < 4; ++mi)
#pragma unroll
      for (int t = 0; t < 4; ++t)
        acc[mi][t] = wmma_bf16(a[mi], b[t], acc[mi][t]);
  }

  // C/D layout: VGPR r -> row (r + 8*hi), lane n -> col n.
  const int n  = lane & 15;
  const int hi = lane >> 4;
#pragma unroll
  for (int mi = 0; mi < 4; ++mi) {
#pragma unroll
    for (int t = 0; t < 4; ++t) {
#pragma unroll
      for (int r = 0; r < 8; ++r) {
        size_t idx = (size_t)(row0 + mi * 16 + hi * 8 + r) * N + col0 + t * 16 + n;
        if (OUT_BF16) ((__bf16*)Cv)[idx] = (__bf16)acc[mi][t][r];
        else          ((float*)Cv)[idx]  = acc[mi][t][r];
      }
    }
  }
}

// ---------------------------------------------------------------------------
// Per-token head-mixing attention:  S = h h^T, P = softmax(S), O = P h
// One token per 256-thread workgroup.  h is [128 heads x 64 dims] bf16.
// h-row staging uses the Tensor Data Mover (one 1-D descriptor per wave).
// ---------------------------------------------------------------------------
__global__ __launch_bounds__(256) void attention_kernel(
    const __bf16* __restrict__ h_bf, __bf16* __restrict__ o_bf) {
  __shared__ __attribute__((aligned(32))) __bf16 h_s[N_HEADS * HEAD_DIM];   // 16 KB
  __shared__ __attribute__((aligned(32))) __bf16 hT_s[HEAD_DIM * N_HEADS];  // 16 KB
  __shared__ __attribute__((aligned(16))) float  S_s[N_HEADS * N_HEADS];    // 64 KB
  __shared__ __attribute__((aligned(32))) __bf16 P_s[N_HEADS * N_HEADS];    // 32 KB

  const int tid    = threadIdx.x;
  const int lane   = tid & 31;
  const int wid    = tid >> 5;                 // 8 waves
  const size_t tok = blockIdx.x;
  const __bf16* hrow = h_bf + tok * (N_HEADS * HEAD_DIM);

  // --- Stage h row (8192 bf16 = 16 KB) into LDS via the TDM. -------------
  // Each wave issues one descriptor for its 2 KB slice (1024 bf16 elements):
  // a 1-D tile, data_size = 2 bytes. TDM ignores EXEC and issues per wave.
  {
    const int ws = __builtin_amdgcn_readfirstlane(wid);   // wave-uniform slice
    const unsigned int SLICE = (N_HEADS * HEAD_DIM) / 8;  // 1024 elements
    unsigned long long ga =
        (unsigned long long)(uintptr_t)hrow + (unsigned long long)ws * SLICE * 2u;
    unsigned int lds = (unsigned int)(uintptr_t)h_s + (unsigned int)ws * SLICE * 2u;

    // D# group 0: [1:0]=count=1; [63:32]=lds_addr; [120:64]=global_addr;
    //             [127:126]=type=2 ("image").
    u32x4 g0;
    g0.x = 1u;
    g0.y = lds;
    g0.z = (unsigned int)ga;
    g0.w = (unsigned int)((ga >> 32) & 0x01FFFFFFull) | (2u << 30);

    // D# group 1: [15:0]=workgroup_mask=0; [17:16]=data_size=1 (2 bytes);
    //             [79:48]=tensor_dim0=1024; [127:112]=tile_dim0=1024;
    //             higher dims / strides zero (1-D copy).
    i32x8 g1;
    g1[0] = (int)(1u << 16);                 // data_size = 2 bytes
    g1[1] = (int)((SLICE & 0xFFFFu) << 16);  // tensor_dim0[15:0] in bits 63:48
    g1[2] = (int)(SLICE >> 16);              // tensor_dim0[31:16] in bits 79:64
    g1[3] = (int)((SLICE & 0xFFFFu) << 16);  // tile_dim0 in bits 127:112
    g1[4] = 0; g1[5] = 0; g1[6] = 0; g1[7] = 0;

    i32x4 gz4 = (i32x4){0, 0, 0, 0};         // groups 2/3: dims unused
    i32x8 gz8 = (i32x8){0, 0, 0, 0, 0, 0, 0, 0};

    // 6-arg toolchain flavor: (g0, g1, g2, g3, g4, cpol).
    __builtin_amdgcn_tensor_load_to_lds(g0, g1, gz4, gz4, gz8, 0);
    __builtin_amdgcn_s_wait_tensorcnt(0);
  }
  __syncthreads();

  // Build transpose hT_s[d][j] = h_s[j][d] (K-contiguous B operand for O=P.h).
  for (int idx = tid; idx < N_HEADS * HEAD_DIM; idx += 256) {
    int j = idx >> 6;          // head
    int d = idx & (HEAD_DIM - 1);
    hT_s[d * N_HEADS + j] = h_s[idx];
  }
  __syncthreads();

  // S = h h^T : 8x8 grid of 16x16 tiles; wave w owns tile-row w. K = 64.
  // A fragments depend only on (wid, k0): hoist out of the jt loop.
  {
    v16bf a0 = load_frag_a(h_s, HEAD_DIM, wid * 16, 0,  lane);
    v16bf a1 = load_frag_a(h_s, HEAD_DIM, wid * 16, 32, lane);
#pragma unroll
    for (int jt = 0; jt < 8; ++jt) {
      v16bf b0 = load_frag_b(h_s, HEAD_DIM, jt * 16, 0,  lane);
      v16bf b1 = load_frag_b(h_s, HEAD_DIM, jt * 16, 32, lane);
      v8f acc = wmma_bf16(a0, b0, (v8f){});
      acc     = wmma_bf16(a1, b1, acc);
      const int n = lane & 15, hi = lane >> 4;
#pragma unroll
      for (int r = 0; r < 8; ++r)
        S_s[(wid * 16 + hi * 8 + r) * N_HEADS + jt * 16 + n] = acc[r];
    }
  }
  __syncthreads();

  // Row softmax in f32 (vectorized f32x4 passes), emit P as bf16.
  if (tid < N_HEADS) {
    f32x4* row4 = (f32x4*)(S_s + tid * N_HEADS);
    f32x4 v = row4[0];
    float m = fmaxf(fmaxf(v.x, v.y), fmaxf(v.z, v.w));
#pragma unroll 4
    for (int j = 1; j < N_HEADS / 4; ++j) {
      v = row4[j];
      m = fmaxf(m, fmaxf(fmaxf(v.x, v.y), fmaxf(v.z, v.w)));
    }
    float sum = 0.0f;
#pragma unroll 4
    for (int j = 0; j < N_HEADS / 4; ++j) {
      f32x4 e = row4[j];
      e.x = __expf(e.x - m); e.y = __expf(e.y - m);
      e.z = __expf(e.z - m); e.w = __expf(e.w - m);
      row4[j] = e;
      sum += e.x + e.y + e.z + e.w;
    }
    float inv = 1.0f / sum;
    bf16x4* prow = (bf16x4*)(P_s + tid * N_HEADS);
#pragma unroll 4
    for (int j = 0; j < N_HEADS / 4; ++j) {
      f32x4 e = row4[j];
      bf16x4 o;
      o.x = (__bf16)(e.x * inv); o.y = (__bf16)(e.y * inv);
      o.z = (__bf16)(e.z * inv); o.w = (__bf16)(e.w * inv);
      prow[j] = o;
    }
  }
  __syncthreads();

  // O = P h : 8x4 grid of 16x16 tiles; wave w owns tile-row w. K = 128.
  // A fragments depend only on (wid, k0): hoist out of the dt loop.
  {
    v16bf a[4];
#pragma unroll
    for (int k = 0; k < 4; ++k)
      a[k] = load_frag_a(P_s, N_HEADS, wid * 16, k * 32, lane);

    __bf16* orow = o_bf + tok * (N_HEADS * HEAD_DIM);
#pragma unroll
    for (int dt = 0; dt < 4; ++dt) {
      v8f acc = (v8f){};
#pragma unroll
      for (int k = 0; k < 4; ++k) {
        v16bf b = load_frag_b(hT_s, N_HEADS, dt * 16, k * 32, lane);
        acc = wmma_bf16(a[k], b, acc);
      }
      const int n = lane & 15, hi = lane >> 4;
#pragma unroll
      for (int r = 0; r < 8; ++r)
        orow[(wid * 16 + hi * 8 + r) * HEAD_DIM + dt * 16 + n] = (__bf16)acc[r];
    }
  }
}

// ---------------------------------------------------------------------------
// Launch
// ---------------------------------------------------------------------------
extern "C" void kernel_launch(void* const* d_in, const int* in_sizes, int n_in,
                              void* d_out, int out_size, void* d_ws, size_t ws_size,
                              hipStream_t stream) {
  const float* x  = (const float*)d_in[0];   // [4,2048,2048]
  const float* w1 = (const float*)d_in[1];   // [8192,2048]
  const float* w2 = (const float*)d_in[2];   // [2048,8192]
  float* out = (float*)d_out;                // [4,2048,2048]

  // Workspace carve-up (all offsets widely aligned; hipMalloc base is aligned).
  char* ws = (char*)d_ws;
  const size_t X_ELEMS  = (size_t)TOKENS * D_MODEL;      // 16,777,216
  const size_t W1_ELEMS = (size_t)D_FF * D_MODEL;        // 16,777,216
  const size_t W2_ELEMS = (size_t)D_MODEL * D_FF;        // 16,777,216
  const size_t H_ELEMS  = (size_t)TOKENS * D_FF;         // 67,108,864
  __bf16* x_bf  = (__bf16*)(ws);
  __bf16* w1_bf = (__bf16*)(ws + 2 * X_ELEMS);
  __bf16* w2_bf = (__bf16*)(ws + 2 * (X_ELEMS + W1_ELEMS));
  __bf16* h_bf  = (__bf16*)(ws + 2 * (X_ELEMS + W1_ELEMS + W2_ELEMS));
  __bf16* o_bf  = (__bf16*)(ws + 2 * (X_ELEMS + W1_ELEMS + W2_ELEMS + H_ELEMS));

  // 1) Precision conversion.
  {
    int n4 = (int)(X_ELEMS / 4);
    int blocks = (n4 + 255) / 256;
    cvt_f32_to_bf16<<<blocks, 256, 0, stream>>>(x,  x_bf,  n4);
    cvt_f32_to_bf16<<<blocks, 256, 0, stream>>>(w1, w1_bf, n4);
    cvt_f32_to_bf16<<<blocks, 256, 0, stream>>>(w2, w2_bf, n4);
  }

  // 2) h = x * w1^T   (M=8192, N=8192, K=2048), bf16 out.
  {
    dim3 grid(D_FF / 128, TOKENS / 256);   // (64, 32)
    gemm_xwT<true><<<grid, 256, 0, stream>>>(x_bf, w1_bf, (void*)h_bf,
                                             TOKENS, D_FF, D_MODEL);
  }

  // 3) Per-token head-mixing attention (8192 workgroups).
  attention_kernel<<<TOKENS, 256, 0, stream>>>(h_bf, o_bf);

  // 4) out = o * w2^T  (M=8192, N=2048, K=8192), f32 out.
  {
    dim3 grid(D_MODEL / 128, TOKENS / 256);  // (16, 32)
    gemm_xwT<false><<<grid, 256, 0, stream>>>(o_bf, w2_bf, (void*)out,
                                              TOKENS, D_MODEL, D_FF);
  }
}
